// AttentionHead_33835752358180
// MI455X (gfx1250) — compile-verified, matrix operand order checked by reading, not run on hardware
//
#include <hip/hip_runtime.h>
#include <hip/hip_bf16.h>

// CDNA5 / gfx1250 attention head:
//   Q=query@Wq+bq; K=key@Wk+bk; V=value@Wv+bv
//   e = exp(Q@K^T / D); weights[i,j] = e[i,j]/rowsum[j]  (reference's broadcast!)
//   out = weights @ V  ==  P @ (V / rowsum[:,None])  with P = e
// GEMMs: v_wmma_f32_16x16x32_bf16, BK=64 (32 WMMAs per LDS fill),
// bf16 tile staging via GLOBAL_LOAD_ASYNC_TO_LDS_B128 where available.

typedef __attribute__((ext_vector_type(16))) __bf16 v16bf;
typedef __attribute__((ext_vector_type(2)))  __bf16 v2bf;
typedef __attribute__((ext_vector_type(8)))  float  v8f;

#define S_DIM 4096
#define D_DIM 1024
#define BK    64
#define LDH   72           // padded LDS stride in halves (144B rows, 16B-aligned frags)

#if defined(__gfx1250__) && __has_builtin(__builtin_amdgcn_global_load_async_to_lds_b128)
#define HAVE_ASYNC_LDS 1
#else
#define HAVE_ASYNC_LDS 0
#endif

// Builtin signature (from hipcc diagnostic): pointee is int __vector(16 bytes)
typedef int v4i __attribute__((vector_size(16)));
typedef __attribute__((address_space(1))) v4i gas_v4i;
typedef __attribute__((address_space(3))) v4i las_v4i;

__device__ __forceinline__ unsigned short f2bf(float x) {
  union { float f; unsigned u; } v; v.f = x;
  unsigned r = v.u + 0x7FFFu + ((v.u >> 16) & 1u);   // round-to-nearest-even
  return (unsigned short)(r >> 16);
}
__device__ __forceinline__ float bf2f(unsigned short h) {
  union { unsigned u; float f; } v; v.u = ((unsigned)h) << 16;
  return v.f;
}
__device__ __forceinline__ unsigned pack2_bf16(float x, float y) {
#if defined(__gfx1250__) && __has_builtin(__builtin_amdgcn_cvt_pk_bf16_f32)
  union { v2bf v; unsigned u; } c;
  c.v = __builtin_amdgcn_cvt_pk_bf16_f32(x, y);      // v_cvt_pk_bf16_f32
  return c.u;
#else
  return (unsigned)f2bf(x) | ((unsigned)f2bf(y) << 16);
#endif
}

// global -> LDS 16-byte copy; async (ASYNCcnt) on gfx1250 toolchains that expose it
__device__ __forceinline__ void cp_b128(void* lds, const void* g) {
#if HAVE_ASYNC_LDS
  __builtin_amdgcn_global_load_async_to_lds_b128(
      (gas_v4i*)(unsigned long long)g,
      (las_v4i*)(unsigned)(unsigned long long)lds,
      0, 0);
#else
  *(uint4*)lds = *(const uint4*)g;
#endif
}
__device__ __forceinline__ void wait_async() {
#if HAVE_ASYNC_LDS
  __builtin_amdgcn_s_wait_asynccnt(0);               // s_wait_asynccnt 0
#endif
}

__device__ __forceinline__ v8f wmma_bf16(v16bf a, v16bf b, v8f c) {
  // (neg_a, A, neg_b, B, c_mod, C, reuse_a, reuse_b)
  return __builtin_amdgcn_wmma_f32_16x16x32_bf16(false, a, false, b, (short)0, c,
                                                 false, false);
}

union Frag { uint4 u[2]; v16bf v; };

// One wave computes a 64x64 tile over a BK=64 LDS tile: 2 k-substeps x 16 WMMAs.
__device__ __forceinline__ void wave_mma(const unsigned short* At,
                                         const unsigned short* Bt,
                                         int wm, int wn, int lane,
                                         v8f acc[4][4]) {
  const int r  = lane & 15;
  const int kb = (lane >> 4) * 8;     // lanes 0-15: K 0..7/16..23; 16-31: 8..15/24..31
#pragma unroll
  for (int ks = 0; ks < 2; ++ks) {
    v16bf af[4], bfr[4];
#pragma unroll
    for (int t = 0; t < 4; ++t) {
      Frag fa, fb;
      const unsigned short* ap = At + (wm + t * 16 + r) * LDH + ks * 32 + kb;
      fa.u[0] = *(const uint4*)ap;          // ds_load_b128
      fa.u[1] = *(const uint4*)(ap + 16);   // ds_load_b128
      af[t] = fa.v;
      const unsigned short* bp = Bt + (wn + t * 16 + r) * LDH + ks * 32 + kb;
      fb.u[0] = *(const uint4*)bp;
      fb.u[1] = *(const uint4*)(bp + 16);
      bfr[t] = fb.v;
    }
#pragma unroll
    for (int mt = 0; mt < 4; ++mt)
#pragma unroll
      for (int nt = 0; nt < 4; ++nt)
        acc[mt][nt] = wmma_bf16(af[mt], bfr[nt], acc[mt][nt]);
  }
}

// ---------------- Projection: Cbf16[M,N] = bf16( A_f32[M,K] @ W_f32[K,N] + bias )
__global__ __launch_bounds__(128)
void proj_gemm(const float* __restrict__ A, const float* __restrict__ W,
               const float* __restrict__ bias, unsigned short* __restrict__ C,
               int M, int N, int K) {
  __shared__ unsigned short At[128 * LDH];
  __shared__ unsigned short Bt[128 * LDH];
  const int tid  = threadIdx.x;
  const int lane = tid & 31, wave = tid >> 5;
  const int wm = (wave >> 1) * 64, wn = (wave & 1) * 64;
  const int m0 = blockIdx.y * 128, n0 = blockIdx.x * 128;
  v8f acc[4][4] = {};

  for (int kk = 0; kk < K; kk += BK) {
    // A tile: 128x64 f32 -> bf16, row-major
#pragma unroll
    for (int i = 0; i < 16; ++i) {
      int pos = tid * 16 + i;
      int row = pos >> 4, c4 = (pos & 15) * 4;
      const float4 v = *(const float4*)(A + (size_t)(m0 + row) * K + kk + c4);
      uint2 p;
      p.x = pack2_bf16(v.x, v.y);
      p.y = pack2_bf16(v.z, v.w);
      *(uint2*)&At[row * LDH + c4] = p;
    }
    // B tile transposed: Bt[n][k] = W[kk+k][n0+n]  (coalesced reads, strided LDS writes)
#pragma unroll
    for (int i = 0; i < 16; ++i) {
      int pos = tid * 16 + i;
      int kr = pos >> 5, c4 = (pos & 31) * 4;
      const float4 v = *(const float4*)(W + (size_t)(kk + kr) * N + n0 + c4);
      Bt[(c4 + 0) * LDH + kr] = f2bf(v.x);
      Bt[(c4 + 1) * LDH + kr] = f2bf(v.y);
      Bt[(c4 + 2) * LDH + kr] = f2bf(v.z);
      Bt[(c4 + 3) * LDH + kr] = f2bf(v.w);
    }
    __syncthreads();
    wave_mma(At, Bt, wm, wn, lane, acc);
    __syncthreads();
  }

  const int nl = lane & 15, mb = (lane >> 4) * 8;
#pragma unroll
  for (int mt = 0; mt < 4; ++mt)
#pragma unroll
    for (int nt = 0; nt < 4; ++nt) {
      int col = n0 + wn + nt * 16 + nl;
      float bv = bias[col];
#pragma unroll
      for (int v = 0; v < 8; ++v) {
        int rowg = m0 + wm + mt * 16 + mb + v;
        C[(size_t)rowg * N + col] = f2bf(acc[mt][nt][v] + bv);
      }
    }
}

// ---------------- Scores: P = exp((Qb @ Kb^T)/D) as bf16; rowsum[i] += sum_j e[i,j]
__global__ __launch_bounds__(128)
void scores_gemm(const unsigned short* __restrict__ Qb,
                 const unsigned short* __restrict__ Kb,
                 unsigned short* __restrict__ P, float* __restrict__ rowsum,
                 int Sn, int Dk) {
  __shared__ unsigned short At[128 * LDH];
  __shared__ unsigned short Bt[128 * LDH];
  const int tid  = threadIdx.x;
  const int lane = tid & 31, wave = tid >> 5;
  const int wm = (wave >> 1) * 64, wn = (wave & 1) * 64;
  const int m0 = blockIdx.y * 128, n0 = blockIdx.x * 128;
  v8f acc[4][4] = {};

  for (int kk = 0; kk < Dk; kk += BK) {
#pragma unroll
    for (int i = 0; i < 8; ++i) {             // A: rows of Qb (async global->LDS)
      int pos = tid * 8 + i;
      int row = pos >> 3, c8 = (pos & 7) * 8;
      cp_b128(&At[row * LDH + c8], Qb + (size_t)(m0 + row) * Dk + kk + c8);
    }
#pragma unroll
    for (int i = 0; i < 8; ++i) {             // B = K^T: rows of Kb are already N-major
      int pos = tid * 8 + i;
      int row = pos >> 3, c8 = (pos & 7) * 8;
      cp_b128(&Bt[row * LDH + c8], Kb + (size_t)(n0 + row) * Dk + kk + c8);
    }
    wait_async();
    __syncthreads();
    wave_mma(At, Bt, wm, wn, lane, acc);
    __syncthreads();
  }

  const float scale = 1.0f / (float)D_DIM;
  const int nl = lane & 15, mb = (lane >> 4) * 8;
#pragma unroll
  for (int mt = 0; mt < 4; ++mt)
#pragma unroll
    for (int v = 0; v < 8; ++v) {
      int rowg = m0 + wm + mt * 16 + mb + v;
      float s = 0.f;
#pragma unroll
      for (int nt = 0; nt < 4; ++nt) {
        int col = n0 + wn + nt * 16 + nl;
        float e = __expf(acc[mt][nt][v] * scale);
        P[(size_t)rowg * Sn + col] = f2bf(e);
        s += e;
      }
      s += __shfl_xor(s, 1, 16);
      s += __shfl_xor(s, 2, 16);
      s += __shfl_xor(s, 4, 16);
      s += __shfl_xor(s, 8, 16);
      if (nl == 0) atomicAdd(&rowsum[rowg], s);   // one atomic per 64 columns
    }
}

// ---------------- Fold reference's column-normalizer into V: Vp[j,:] = Vb[j,:]/rowsum[j]
__global__ void scale_v(const unsigned short* __restrict__ Vb,
                        const float* __restrict__ rowsum,
                        unsigned short* __restrict__ Vp, int Dn) {
  const int row = blockIdx.x;
  const float inv = 1.0f / rowsum[row];
  const unsigned short* src = Vb + (size_t)row * Dn;
  unsigned short* dst = Vp + (size_t)row * Dn;
  for (int i = threadIdx.x; i < Dn; i += blockDim.x)
    dst[i] = f2bf(bf2f(src[i]) * inv);
}

// ---------------- Output: out_f32[S,D] = P_bf16[S,S] @ Vp_bf16[S,D]
__global__ __launch_bounds__(128)
void out_gemm(const unsigned short* __restrict__ P,
              const unsigned short* __restrict__ Vp,
              float* __restrict__ out, int Sn, int Dn) {
  __shared__ unsigned short At[128 * LDH];
  __shared__ unsigned short Bt[128 * LDH];
  const int tid  = threadIdx.x;
  const int lane = tid & 31, wave = tid >> 5;
  const int wm = (wave >> 1) * 64, wn = (wave & 1) * 64;
  const int m0 = blockIdx.y * 128, n0 = blockIdx.x * 128;
  v8f acc[4][4] = {};

  for (int kk = 0; kk < Sn; kk += BK) {
#pragma unroll
    for (int i = 0; i < 8; ++i) {             // A: rows of P (async global->LDS)
      int pos = tid * 8 + i;
      int row = pos >> 3, c8 = (pos & 7) * 8;
      cp_b128(&At[row * LDH + c8], P + (size_t)(m0 + row) * Sn + kk + c8);
    }
#pragma unroll
    for (int i = 0; i < 8; ++i) {             // B transposed: Bt[n][k] = Vp[kk+k][n0+n]
      int pos = tid * 8 + i;
      int kr = pos >> 4, c8 = (pos & 15) * 8;
      union { uint4 u; unsigned short h[8]; } v;
      v.u = *(const uint4*)(Vp + (size_t)(kk + kr) * Dn + n0 + c8);
#pragma unroll
      for (int e = 0; e < 8; ++e) Bt[(c8 + e) * LDH + kr] = v.h[e];
    }
    wait_async();
    __syncthreads();
    wave_mma(At, Bt, wm, wn, lane, acc);
    __syncthreads();
  }

  const int nl = lane & 15, mb = (lane >> 4) * 8;
#pragma unroll
  for (int mt = 0; mt < 4; ++mt)
#pragma unroll
    for (int nt = 0; nt < 4; ++nt) {
      int col = n0 + wn + nt * 16 + nl;
#pragma unroll
      for (int v = 0; v < 8; ++v) {
        int rowg = m0 + wm + mt * 16 + mb + v;
        out[(size_t)rowg * Dn + col] = acc[mt][nt][v];
      }
    }
}

extern "C" void kernel_launch(void* const* d_in, const int* in_sizes, int n_in,
                              void* d_out, int out_size, void* d_ws, size_t ws_size,
                              hipStream_t stream) {
  (void)in_sizes; (void)n_in; (void)out_size; (void)ws_size;
  const float* query = (const float*)d_in[0];
  const float* key   = (const float*)d_in[1];
  const float* value = (const float*)d_in[2];
  const float* Wq    = (const float*)d_in[3];
  const float* bq    = (const float*)d_in[4];
  const float* Wk    = (const float*)d_in[5];
  const float* bk    = (const float*)d_in[6];
  const float* Wv    = (const float*)d_in[7];
  const float* bv    = (const float*)d_in[8];

  const int S = S_DIM, D = D_DIM;
  // Workspace: Qb|Kb|Vb|Vp (bf16, S*D each), P (bf16, S*S), rowsum (f32, S)
  unsigned short* Qb = (unsigned short*)d_ws;
  unsigned short* Kb = Qb + (size_t)S * D;
  unsigned short* Vb = Kb + (size_t)S * D;
  unsigned short* Vp = Vb + (size_t)S * D;
  unsigned short* P  = Vp + (size_t)S * D;
  float* rowsum      = (float*)(P + (size_t)S * S);

  dim3 blk(128);
  dim3 gproj(D / 128, S / 128);   // 8 x 32
  proj_gemm<<<gproj, blk, 0, stream>>>(query, Wq, bq, Qb, S, D, D);
  proj_gemm<<<gproj, blk, 0, stream>>>(key,   Wk, bk, Kb, S, D, D);
  proj_gemm<<<gproj, blk, 0, stream>>>(value, Wv, bv, Vb, S, D, D);

  (void)hipMemsetAsync(rowsum, 0, (size_t)S * sizeof(float), stream);

  dim3 gsc(S / 128, S / 128);     // 32 x 32
  scores_gemm<<<gsc, blk, 0, stream>>>(Qb, Kb, P, rowsum, S, D);

  scale_v<<<dim3(S), dim3(256), 0, stream>>>(Vb, rowsum, Vp, D);

  out_gemm<<<dim3(D / 128, S / 128), blk, 0, stream>>>(P, Vp, (float*)d_out, S, D);
}
